// MultiHeadSelfAttention_33638183862336
// MI455X (gfx1250) — compile-verified, matrix operand order checked by reading, not run on hardware
//
#include <hip/hip_runtime.h>

#define B_  4
#define S_  2048
#define D_  1024
#define H_  16
#define PD_ 64

typedef __attribute__((ext_vector_type(16))) __bf16 v16bf;
typedef __attribute__((ext_vector_type(8)))  float  v8f;
typedef int v4i_b128 __attribute__((vector_size(16)));   // async b128 payload type

// float -> bf16 (round to nearest even), as raw bits
__device__ __forceinline__ unsigned f2bf_bits(float f) {
    unsigned u = __builtin_bit_cast(unsigned, f);
    unsigned r = 0x7FFFu + ((u >> 16) & 1u);
    return (u + r) >> 16;
}
__device__ __forceinline__ unsigned pack2bf(float lo, float hi) {
    return f2bf_bits(lo) | (f2bf_bits(hi) << 16);
}
__device__ __forceinline__ __bf16 f2bf(float f) {
    return __builtin_bit_cast(__bf16, (unsigned short)f2bf_bits(f));
}

__device__ __forceinline__ v8f wmma_bf16(v16bf a, v16bf b, v8f c) {
    // D = A(16x32 bf16) * B(32x16 bf16) + C(16x16 f32)
    return __builtin_amdgcn_wmma_f32_16x16x32_bf16(
        false, a, false, b, (short)0, c, false, false);
}

// ---- gfx1250 async global->LDS copy (ASYNCcnt path), guarded ---------------
#if defined(__gfx1250__) && __has_builtin(__builtin_amdgcn_global_load_async_to_lds_b128)
#define HAVE_ASYNC_LDS 1
#else
#define HAVE_ASYNC_LDS 0
#endif

__device__ __forceinline__ void copy16_g2l(const void* gsrc, void* ldst) {
#if HAVE_ASYNC_LDS
    __builtin_amdgcn_global_load_async_to_lds_b128(
        (__attribute__((address_space(1))) v4i_b128*)gsrc,
        (__attribute__((address_space(3))) v4i_b128*)ldst, 0, 0);
#else
    *reinterpret_cast<uint4*>(ldst) = *reinterpret_cast<const uint4*>(gsrc);
#endif
}
__device__ __forceinline__ void async_wait0() {
#if HAVE_ASYNC_LDS
#if __has_builtin(__builtin_amdgcn_s_wait_asynccnt)
    __builtin_amdgcn_s_wait_asynccnt(0);
#else
    asm volatile("s_wait_asynccnt 0" ::: "memory");
#endif
#endif
}

// ---------------------------------------------------------------------------
// Kernel 1: QKV projection.  X[B*S, D] @ W[D, D] + bias -> bf16 [B, H, S, PD]
// One 64x64 output tile per 256-thread (8 wave32) block; 2 subtiles per wave.
// ---------------------------------------------------------------------------
__global__ __launch_bounds__(256)
void qkv_proj(const float* __restrict__ qx, const float* __restrict__ kx,
              const float* __restrict__ vx,
              const float* __restrict__ Wq, const float* __restrict__ bq,
              const float* __restrict__ Wk, const float* __restrict__ bk,
              const float* __restrict__ Wv, const float* __restrict__ bv,
              unsigned short* __restrict__ Qp, unsigned short* __restrict__ Kp,
              unsigned short* __restrict__ Vp)
{
    const int z = blockIdx.z;
    const float* X  = (z == 0) ? qx : (z == 1) ? kx : vx;
    const float* W  = (z == 0) ? Wq : (z == 1) ? Wk : Wv;
    const float* bs = (z == 0) ? bq : (z == 1) ? bk : bv;
    unsigned short* O = (z == 0) ? Qp : (z == 1) ? Kp : Vp;

    __shared__ __align__(32) unsigned short As[64][48];  // 64 rows x 32 k (pitch 96B)
    __shared__ __align__(32) unsigned short Bt[64][48];  // 64 cols x 32 k (transposed W)

    const int row0 = blockIdx.x * 64;       // row in [0, B*S)
    const int col0 = blockIdx.y * 64;       // col in [0, D)
    const int tid  = threadIdx.x;
    const int wave = tid >> 5;
    const int lane = tid & 31;
    const int n    = lane & 15;             // fragment column / row index
    const int g    = lane >> 4;             // half-wave group (k-half select)

    const int st0 = wave * 2, st1 = wave * 2 + 1;       // subtile ids in 4x4 grid
    const int r0 = (st0 >> 2) * 16, c0 = (st0 & 3) * 16;
    const int r1 = (st1 >> 2) * 16, c1 = (st1 & 3) * 16;

    v8f acc0 = {}; v8f acc1 = {};

    for (int k0 = 0; k0 < D_; k0 += 32) {
        __syncthreads();
        // stage A tile (f32 -> bf16): 512 float4 chunks, 2 per thread
        #pragma unroll
        for (int it = 0; it < 2; ++it) {
            int e = tid + it * 256;
            int r = e >> 3, c4 = (e & 7) * 4;
            float4 f = *reinterpret_cast<const float4*>(&X[(size_t)(row0 + r) * D_ + k0 + c4]);
            uint2 p; p.x = pack2bf(f.x, f.y); p.y = pack2bf(f.z, f.w);
            *reinterpret_cast<uint2*>(&As[r][c4]) = p;
        }
        // stage W tile transposed: Bt[c][kk] = W[k0+kk][col0+c]; coalesced float4 reads
        #pragma unroll
        for (int it = 0; it < 2; ++it) {
            int e = tid + it * 256;
            int kk = e >> 4, c4 = (e & 15) * 4;
            float4 f = *reinterpret_cast<const float4*>(&W[(size_t)(k0 + kk) * D_ + col0 + c4]);
            Bt[c4 + 0][kk] = (unsigned short)f2bf_bits(f.x);
            Bt[c4 + 1][kk] = (unsigned short)f2bf_bits(f.y);
            Bt[c4 + 2][kk] = (unsigned short)f2bf_bits(f.z);
            Bt[c4 + 3][kk] = (unsigned short)f2bf_bits(f.w);
        }
        if (k0 + 32 < D_) {                 // global_prefetch_b8 of next tiles
            __builtin_prefetch(&X[(size_t)(row0 + (tid >> 2)) * D_ + k0 + 32], 0, 1);
            __builtin_prefetch(&W[(size_t)(k0 + 32 + (tid & 31)) * D_ + col0], 0, 1);
        }
        __syncthreads();

        const int kh = g * 16;
        const v16bf a0 = *reinterpret_cast<const v16bf*>(&As[r0 + n][kh]);
        const v16bf b0 = *reinterpret_cast<const v16bf*>(&Bt[c0 + n][kh]);
        acc0 = wmma_bf16(a0, b0, acc0);
        const v16bf a1 = *reinterpret_cast<const v16bf*>(&As[r1 + n][kh]);
        const v16bf b1 = *reinterpret_cast<const v16bf*>(&Bt[c1 + n][kh]);
        acc1 = wmma_bf16(a1, b1, acc1);
    }

    // C layout: lane holds column n; rows m = g*8 + i.  Scatter to [B,H,S,PD].
    auto store_tile = [&](const v8f& acc, int rr, int cc) {
        const int gcol = col0 + cc + n;
        const int h = gcol >> 6, pd = gcol & 63;
        const float bias = bs[gcol];
        #pragma unroll
        for (int i = 0; i < 8; ++i) {
            int grow = row0 + rr + g * 8 + i;            // b*S + s
            int b = grow >> 11, s = grow & (S_ - 1);
            size_t idx = (((size_t)(b * H_ + h)) * S_ + s) * PD_ + pd;
            O[idx] = (unsigned short)f2bf_bits(acc[i] + bias);
        }
    };
    store_tile(acc0, r0, c0);
    store_tile(acc1, r1, c1);
}

// ---------------------------------------------------------------------------
// Kernel 2: flash attention per (b, h).  Each wave owns a 16-query strip,
// block covers 128 queries; iterates all 32 KV tiles of 64 with online
// softmax.  S^T = K * Q^T  (softmax dim = accumulator M),  O^T = V^T * P.
// ---------------------------------------------------------------------------
__global__ __launch_bounds__(256)
void attention(const unsigned short* __restrict__ Qp,
               const unsigned short* __restrict__ Kp,
               const unsigned short* __restrict__ Vp,
               const int* __restrict__ q_mask, const int* __restrict__ kv_mask,
               unsigned short* __restrict__ attn)
{
    __shared__ __align__(32) unsigned short Qs[128][80];  // 128 queries x PD
    __shared__ __align__(32) unsigned short Ks[64][80];   // kv tile x PD
    __shared__ __align__(32) unsigned short Vts[64][80];  // PD x kv tile (V^T)

    const int b = blockIdx.z, h = blockIdx.y;
    const int q0 = blockIdx.x * 128;
    const int tid = threadIdx.x, wave = tid >> 5, lane = tid & 31;
    const int n = lane & 15, g = lane >> 4;
    const size_t head = ((size_t)(b * H_ + h)) * S_ * PD_;

    // stage the 128x64 Q block once: 1024 16B chunks, 4 per thread (async path)
    #pragma unroll
    for (int it = 0; it < 4; ++it) {
        int ch = tid + it * 256;
        int r = ch >> 3, c8 = (ch & 7) * 8;
        copy16_g2l(&Qp[head + (size_t)(q0 + r) * PD_ + c8], &Qs[r][c8]);
    }

    const int gq = q0 + wave * 16 + n;          // this lane's query (column n)
    float m_run = -1.0e30f, l_run = 0.0f;
    v8f Ov[4] = {{}, {}, {}, {}};               // O^T: 4 pd-subtiles x 16 q

    for (int t = 0; t < S_ / 64; ++t) {
        const int kv0 = t * 64;
        async_wait0();
        __syncthreads();
        // stage K tile: 512 16B chunks, 2 per thread (async path)
        #pragma unroll
        for (int it = 0; it < 2; ++it) {
            int ch = tid + it * 256;
            int r = ch >> 3, c8 = (ch & 7) * 8;
            copy16_g2l(&Kp[head + (size_t)(kv0 + r) * PD_ + c8], &Ks[r][c8]);
        }
        // stage V transposed: vector global reads, scalar transposed LDS stores
        #pragma unroll
        for (int it = 0; it < 2; ++it) {
            int ch = tid + it * 256;
            int kvl = ch >> 3, p8 = (ch & 7) * 8;
            union { uint4 u; unsigned short s[8]; } w;
            w.u = *reinterpret_cast<const uint4*>(&Vp[head + (size_t)(kv0 + kvl) * PD_ + p8]);
            #pragma unroll
            for (int j = 0; j < 8; ++j) Vts[p8 + j][kvl] = w.s[j];
        }
        async_wait0();
        __syncthreads();

        // S^T tile: 4 kv-subtiles (M) x 16 queries (N), K-loop over PD=64
        v8f sacc[4];
        #pragma unroll
        for (int ms = 0; ms < 4; ++ms) {
            v8f acc = {};
            #pragma unroll
            for (int ks = 0; ks < 2; ++ks) {
                const v16bf af = *reinterpret_cast<const v16bf*>(&Ks[ms * 16 + n][ks * 32 + g * 16]);
                const v16bf bf = *reinterpret_cast<const v16bf*>(&Qs[wave * 16 + n][ks * 32 + g * 16]);
                acc = wmma_bf16(af, bf, acc);
            }
            sacc[ms] = acc;
        }

        // scale + faithful mask (subtract 1e-9 where dropped), then online softmax
        float tmax = -1.0e30f;
        #pragma unroll
        for (int ms = 0; ms < 4; ++ms)
            #pragma unroll
            for (int i = 0; i < 8; ++i) {
                int gkv = kv0 + ms * 16 + g * 8 + i;
                float sv = sacc[ms][i] * 0.125f;          // 1/sqrt(64)
                bool keep = (kv_mask[b * S_ + gkv] != 0) && (gkv <= gq);
                sv -= keep ? 0.0f : 1e-9f;
                sacc[ms][i] = sv;
                tmax = fmaxf(tmax, sv);
            }
        tmax = fmaxf(tmax, __shfl_xor(tmax, 16));
        const float mnew = fmaxf(m_run, tmax);
        const float corr = __expf(m_run - mnew);
        float tsum = 0.0f;
        #pragma unroll
        for (int ms = 0; ms < 4; ++ms)
            #pragma unroll
            for (int i = 0; i < 8; ++i) {
                float p = __expf(sacc[ms][i] - mnew);
                sacc[ms][i] = p;
                tsum += p;
            }
        tsum += __shfl_xor(tsum, 16);
        l_run = l_run * corr + tsum;
        m_run = mnew;
        #pragma unroll
        for (int ps = 0; ps < 4; ++ps)
            #pragma unroll
            for (int i = 0; i < 8; ++i) Ov[ps][i] *= corr;

        // O^T += V^T * P : rebuild P as bf16 B-fragments via half-wave shuffles
        #pragma unroll
        for (int ks2 = 0; ks2 < 2; ++ks2) {
            const v8f pa = sacc[ks2 * 2 + 0];
            const v8f pb = sacc[ks2 * 2 + 1];
            v16bf pf;
            #pragma unroll
            for (int i = 0; i < 8; ++i) {
                float sa = __shfl_xor(pa[i], 16);
                float sb = __shfl_xor(pb[i], 16);
                float lo = (g == 0) ? pa[i] : sb;   // rows 0..7 of this group's subtile
                float hi = (g == 0) ? sa : pb[i];   // rows 8..15
                pf[i]     = f2bf(lo);
                pf[8 + i] = f2bf(hi);
            }
            #pragma unroll
            for (int ps = 0; ps < 4; ++ps) {
                const v16bf vf = *reinterpret_cast<const v16bf*>(&Vts[ps * 16 + n][ks2 * 32 + g * 16]);
                Ov[ps] = wmma_bf16(vf, pf, Ov[ps]);
            }
        }
    }

    // finalize: divide by softmax sum, apply query mask, write bf16 [B,S,D]
    const float qscale = (q_mask[b * S_ + gq] != 0) ? (1.0f / l_run) : 0.0f;
    #pragma unroll
    for (int ps = 0; ps < 4; ++ps)
        #pragma unroll
        for (int i = 0; i < 8; ++i) {
            int pd = ps * 16 + g * 8 + i;
            attn[((size_t)b * S_ + gq) * D_ + h * PD_ + pd] =
                (unsigned short)f2bf_bits(Ov[ps][i] * qscale);
        }
}

// ---------------------------------------------------------------------------
// Kernel 3: output projection.  attn(bf16)[B*S, D] @ Wo + bo -> f32 [B,S,D]
// ---------------------------------------------------------------------------
__global__ __launch_bounds__(256)
void out_proj(const unsigned short* __restrict__ A, const float* __restrict__ Wo,
              const float* __restrict__ bo, float* __restrict__ out)
{
    __shared__ __align__(32) unsigned short As[64][48];
    __shared__ __align__(32) unsigned short Bt[64][48];

    const int row0 = blockIdx.x * 64;
    const int col0 = blockIdx.y * 64;
    const int tid  = threadIdx.x;
    const int wave = tid >> 5;
    const int lane = tid & 31;
    const int n = lane & 15, g = lane >> 4;

    const int st0 = wave * 2, st1 = wave * 2 + 1;
    const int r0 = (st0 >> 2) * 16, c0 = (st0 & 3) * 16;
    const int r1 = (st1 >> 2) * 16, c1 = (st1 & 3) * 16;

    v8f acc0 = {}; v8f acc1 = {};

    for (int k0 = 0; k0 < D_; k0 += 32) {
        __syncthreads();
        // stage A tile: bf16 verbatim, 256 16B chunks, 1 per thread (async path)
        {
            int r = tid >> 2, c8 = (tid & 3) * 8;
            copy16_g2l(&A[(size_t)(row0 + r) * D_ + k0 + c8], &As[r][c8]);
        }
        // stage Wo tile transposed (f32 -> bf16)
        #pragma unroll
        for (int it = 0; it < 2; ++it) {
            int e = tid + it * 256;
            int kk = e >> 4, c4 = (e & 15) * 4;
            float4 f = *reinterpret_cast<const float4*>(&Wo[(size_t)(k0 + kk) * D_ + col0 + c4]);
            Bt[c4 + 0][kk] = (unsigned short)f2bf_bits(f.x);
            Bt[c4 + 1][kk] = (unsigned short)f2bf_bits(f.y);
            Bt[c4 + 2][kk] = (unsigned short)f2bf_bits(f.z);
            Bt[c4 + 3][kk] = (unsigned short)f2bf_bits(f.w);
        }
        if (k0 + 32 < D_)
            __builtin_prefetch(&Wo[(size_t)(k0 + 32 + (tid & 31)) * D_ + col0], 0, 1);
        async_wait0();
        __syncthreads();

        const int kh = g * 16;
        const v16bf a0 = *reinterpret_cast<const v16bf*>(&As[r0 + n][kh]);
        const v16bf b0 = *reinterpret_cast<const v16bf*>(&Bt[c0 + n][kh]);
        acc0 = wmma_bf16(a0, b0, acc0);
        const v16bf a1 = *reinterpret_cast<const v16bf*>(&As[r1 + n][kh]);
        const v16bf b1 = *reinterpret_cast<const v16bf*>(&Bt[c1 + n][kh]);
        acc1 = wmma_bf16(a1, b1, acc1);
    }

    auto store_tile = [&](const v8f& acc, int rr, int cc) {
        const int gcol = col0 + cc + n;
        const float bias = bo[gcol];
        #pragma unroll
        for (int i = 0; i < 8; ++i) {
            int grow = row0 + rr + g * 8 + i;
            out[(size_t)grow * D_ + gcol] = acc[i] + bias;
        }
    };
    store_tile(acc0, r0, c0);
    store_tile(acc1, r1, c1);
}

// ---------------------------------------------------------------------------
extern "C" void kernel_launch(void* const* d_in, const int* in_sizes, int n_in,
                              void* d_out, int out_size, void* d_ws, size_t ws_size,
                              hipStream_t stream)
{
    (void)in_sizes; (void)n_in; (void)out_size; (void)ws_size;
    const float* q   = (const float*)d_in[0];
    const float* k   = (const float*)d_in[1];
    const float* v   = (const float*)d_in[2];
    const int*   qm  = (const int*)d_in[3];
    const int*   kvm = (const int*)d_in[4];
    const float* Wq  = (const float*)d_in[5];
    const float* bq  = (const float*)d_in[6];
    const float* Wk  = (const float*)d_in[7];
    const float* bk  = (const float*)d_in[8];
    const float* Wv  = (const float*)d_in[9];
    const float* bv  = (const float*)d_in[10];
    const float* Wo  = (const float*)d_in[11];
    const float* bo  = (const float*)d_in[12];
    float* out = (float*)d_out;

    const size_t headElems = (size_t)B_ * H_ * S_ * PD_;   // 8.4M bf16 each
    unsigned short* Qp   = (unsigned short*)d_ws;
    unsigned short* Kp   = Qp + headElems;
    unsigned short* Vp   = Kp + headElems;
    unsigned short* attn = Vp + headElems;

    dim3 g1(B_ * S_ / 64, D_ / 64, 3);
    qkv_proj<<<g1, 256, 0, stream>>>(q, k, v, Wq, bq, Wk, bk, Wv, bv, Qp, Kp, Vp);

    dim3 g2(S_ / 128, H_, B_);
    attention<<<g2, 256, 0, stream>>>(Qp, Kp, Vp, qm, kvm, attn);

    dim3 g3(B_ * S_ / 64, D_ / 64, 1);
    out_proj<<<g3, 256, 0, stream>>>(attn, Wo, bo, out);
}
